// NystromAttention_77051713290438
// MI455X (gfx1250) — compile-verified, matrix operand order checked by reading, not run on hardware
//
#include <hip/hip_runtime.h>

// Nystrom attention for MI455X (gfx1250), wave32 + WMMA bf16 (f32 accum).
// Shapes: b=4, n=8192, dim=512, h=8, dh=64, m=256 landmarks, l=32.
// Strategy: all large GEMMs -> v_wmma_f32_16x16x32_bf16; attn1/attn3 are
// never materialized (flash-style fused softmax GEMMs, P re-layout via LDS);
// K/V panels in the flash kernels are staged global->LDS with the gfx1250
// async-DMA path (global_load_async_to_lds_b128 + s_wait_asynccnt);
// Newton-Schulz pinv folds (cI - X) into the GEMM epilogue.
// Workspace requirement: ~250 MB.

#define B_     4
#define N_     8192
#define DIM_   512
#define H_     8
#define DH_    64
#define M_     256
#define BH_    32           // B_*H_
#define ROWS_  32768        // B_*N_
#define INNER_ 512          // H_*DH_
#define KQKV_  1536

typedef __bf16 bf16_t;
typedef __attribute__((ext_vector_type(16))) __bf16 v16bf;
typedef __attribute__((ext_vector_type(8)))  __bf16 bfx8;
typedef __attribute__((ext_vector_type(8)))  float  v8f;
typedef __attribute__((ext_vector_type(4)))  float  f32x4;

// ---------------------------------------------------------------- WMMA core
__device__ __forceinline__ v8f wmma_bf16(v16bf a, v16bf b, v8f c) {
  return __builtin_amdgcn_wmma_f32_16x16x32_bf16(false, a, false, b,
                                                 (short)0, c, false, false);
}

// gfx1250 async global->LDS DMA (ASYNCcnt-tracked).  Inline asm: portable
// across ROCm 7.2 / amdgpu-toolchain per the CDNA5 bridge doc.
__device__ __forceinline__ void async_ld_b128(unsigned lds_off, const void* gaddr) {
  asm volatile("global_load_async_to_lds_b128 %0, %1, off"
               :: "v"(lds_off), "v"(gaddr)
               : "memory");
}
__device__ __forceinline__ void wait_async0() {
  asm volatile("s_wait_asynccnt 0x0" ::: "memory");
}

// A-fragment (16x32, row per lane&15): element j holds K = (j&7)+8*half+16*(j>>3)
// -> two contiguous 8-element (16B) chunks at +half*8 and +16+half*8.
__device__ __forceinline__ v16bf frag_contig(const bf16_t* p) {
  bfx8 c0 = *(const bfx8*)(p);
  bfx8 c1 = *(const bfx8*)(p + 16);
  v16bf a;
#pragma unroll
  for (int j = 0; j < 8; ++j) { a[j] = c0[j]; a[8 + j] = c1[j]; }
  return a;
}
__device__ __forceinline__ v16bf frag_contig(const float* p) {
  f32x4 u0 = *(const f32x4*)(p + 0);
  f32x4 u1 = *(const f32x4*)(p + 4);
  f32x4 u2 = *(const f32x4*)(p + 16);
  f32x4 u3 = *(const f32x4*)(p + 20);
  v16bf a;
#pragma unroll
  for (int j = 0; j < 4; ++j) {
    a[j]      = (bf16_t)u0[j];  a[4 + j]  = (bf16_t)u1[j];
    a[8 + j]  = (bf16_t)u2[j];  a[12 + j] = (bf16_t)u3[j];
  }
  return a;
}

// B-fragment (32x16, col per lane&15) from row-major [K x N]: strided gather
// (lanes 0..15 of each half read 16 consecutive elements -> coalesced across
// the wave even though per-lane scalar).
template <typename T>
__device__ __forceinline__ v16bf frag_b_gather(const T* Bp, int ldb, int col, int half) {
  v16bf b;
#pragma unroll
  for (int j = 0; j < 16; ++j) {
    int k = (j & 7) + (half << 3) + ((j >> 3) << 4);
    b[j] = (bf16_t)Bp[(long)k * ldb + col];
  }
  return b;
}

// ---------------------------------------------- generic batched WMMA GEMM
// C[m,n] = mul * sum_k A[m,k]*B[k,n]  (+ diag on m==n) (+ bias[n])
// TRANSB=true: B given as [N x K] row-major (contiguous fast path).
// One wave per 16x64 output tile. Requires M%16==0, N%64==0, K%32==0.
template <typename TA, typename TB, bool TRANSB, bool BIAS>
__global__ void gemm_wmma(const TA* __restrict__ A, const TB* __restrict__ Bm,
                          float* __restrict__ C, int M, int N, int K,
                          int lda, int ldb, int ldc,
                          long sA, long sB, long sC,
                          float mul, float diag, const float* __restrict__ bias) {
  const int lane = threadIdx.x & 31;
  const int half = lane >> 4;
  const int ln   = lane & 15;
  const int n0 = blockIdx.x * 64;
  const int m0 = blockIdx.y * 16;
  A  += (long)blockIdx.z * sA;
  Bm += (long)blockIdx.z * sB;
  C  += (long)blockIdx.z * sC;
  v8f acc[4] = {};
  const TA* arow = A + (long)(m0 + ln) * lda;
  for (int k0 = 0; k0 < K; k0 += 32) {
    if (k0 + 32 < K) __builtin_prefetch(arow + k0 + 32, 0, 0);  // global_prefetch_b8
    v16bf a = frag_contig(arow + k0 + half * 8);
#pragma unroll
    for (int t = 0; t < 4; ++t) {
      v16bf b;
      if (TRANSB)
        b = frag_contig(Bm + (long)(n0 + t * 16 + ln) * ldb + k0 + half * 8);
      else
        b = frag_b_gather(Bm + (long)k0 * ldb, ldb, n0 + t * 16 + ln, half);
      acc[t] = wmma_bf16(a, b, acc[t]);
    }
  }
#pragma unroll
  for (int t = 0; t < 4; ++t) {
    const int n = n0 + t * 16 + ln;
#pragma unroll
    for (int r = 0; r < 8; ++r) {
      const int m = m0 + r + half * 8;
      float v = mul * acc[t][r];
      if (m == n) v += diag;
      if (BIAS) v += bias[n];
      C[(long)m * ldc + n] = v;
    }
  }
}

// ------------------------------------- qkv projection with head scatter
// X[32768,512] f32 @ wqkvT[1536,512] bf16 -> q/k/v bf16 [b,h,n,dh], q*0.125
__global__ void k_qkv_gemm(const float* __restrict__ X, const bf16_t* __restrict__ WT,
                           bf16_t* __restrict__ Qb, bf16_t* __restrict__ Kb,
                           bf16_t* __restrict__ Vb) {
  const int lane = threadIdx.x & 31, half = lane >> 4, ln = lane & 15;
  const int n0 = blockIdx.x * 64;
  const int m0 = blockIdx.y * 16;
  v8f acc[4] = {};
  const float* xrow = X + (long)(m0 + ln) * DIM_;
  for (int k0 = 0; k0 < DIM_; k0 += 32) {
    if (k0 + 32 < DIM_) __builtin_prefetch(xrow + k0 + 32, 0, 0);
    v16bf a = frag_contig(xrow + k0 + half * 8);
#pragma unroll
    for (int t = 0; t < 4; ++t) {
      v16bf b = frag_contig(WT + (long)(n0 + t * 16 + ln) * DIM_ + k0 + half * 8);
      acc[t] = wmma_bf16(a, b, acc[t]);
    }
  }
#pragma unroll
  for (int t = 0; t < 4; ++t) {
    const int c = n0 + t * 16 + ln;
    const int which = c >> 9, inner = c & 511, h = inner >> 6, d = inner & 63;
#pragma unroll
    for (int r = 0; r < 8; ++r) {
      const int row = m0 + r + half * 8;
      const int bb = row >> 13, nidx = row & (N_ - 1);
      const long dst = (((long)bb * H_ + h) * N_ + nidx) * DH_ + d;
      float v = acc[t][r];
      if (which == 0)      Qb[dst] = (bf16_t)(v * 0.125f);  // dh^-0.5
      else if (which == 1) Kb[dst] = (bf16_t)v;
      else                 Vb[dst] = (bf16_t)v;
    }
  }
}

// ------------------------------------------------ landmark means (l=32)
__global__ void k_landmarks(const bf16_t* __restrict__ Qb, const bf16_t* __restrict__ Kb,
                            bf16_t* __restrict__ Ql, bf16_t* __restrict__ Kl) {
  long i = (long)blockIdx.x * blockDim.x + threadIdx.x;
  if (i >= (long)BH_ * M_ * DH_) return;
  const int d = (int)(i & 63);
  const int mi = (int)((i >> 6) & 255);
  const long bh = i >> 14;
  const bf16_t* q = Qb + (bh * N_ + (long)mi * 32) * DH_ + d;
  const bf16_t* k = Kb + (bh * N_ + (long)mi * 32) * DH_ + d;
  float sq = 0.f, sk = 0.f;
#pragma unroll
  for (int t = 0; t < 32; ++t) { sq += (float)q[t * DH_]; sk += (float)k[t * DH_]; }
  Ql[i] = (bf16_t)(sq * (1.f / 32.f));
  Kl[i] = (bf16_t)(sk * (1.f / 32.f));
}

// ------------------- fused softmax(Q K^T) V, exact online softmax (flash)
// One wave handles 16 query rows; streams keys in chunks of 32.
// K/V chunks (4KB each) are DMA'd global->LDS via async b128 (512B per issue
// across the wave, no VGPR round-trip), then fragments are served from LDS.
// Q[bh,nq,64], K[bh,nk,64], V[bh,nk,64] bf16 -> O[bh,nq,64] f32.
__global__ void k_flash16(const bf16_t* __restrict__ Q, const bf16_t* __restrict__ Kt,
                          const bf16_t* __restrict__ V, float* __restrict__ O,
                          int nq, int nk) {
  __shared__ __align__(16) bf16_t Ks[32 * DH_];   // 4KB key chunk
  __shared__ __align__(16) bf16_t Vs[32 * DH_];   // 4KB value chunk
  __shared__ __align__(16) bf16_t Pl[16 * 32];    // P D->A re-layout buffer
  const int lane = threadIdx.x & 31, half = lane >> 4, ln = lane & 15;
  const long bh = blockIdx.y;
  const bf16_t* q = Q + bh * (long)nq * DH_;
  const bf16_t* k = Kt + bh * (long)nk * DH_;
  const bf16_t* v = V + bh * (long)nk * DH_;
  float* out = O + bh * (long)nq * DH_;
  const int m0 = blockIdx.x * 16;
  const v16bf aq0 = frag_contig(q + (long)(m0 + ln) * DH_ + half * 8);
  const v16bf aq1 = frag_contig(q + (long)(m0 + ln) * DH_ + 32 + half * 8);
  const unsigned lK0 = (unsigned)(unsigned long long)(const void*)Ks + lane * 16;
  const unsigned lV0 = (unsigned)(unsigned long long)(const void*)Vs + lane * 16;
  float mrow[8], lrow[8];
  v8f o[4] = {};
#pragma unroll
  for (int r = 0; r < 8; ++r) { mrow[r] = -1e30f; lrow[r] = 0.f; }
  for (int c0 = 0; c0 < nk; c0 += 32) {
    // ---- async DMA of this chunk's K and V panels into LDS
    {
      const char* gK = (const char*)(k + (long)c0 * DH_) + lane * 16;
      const char* gV = (const char*)(v + (long)c0 * DH_) + lane * 16;
#pragma unroll
      for (int i = 0; i < 8; ++i) {
        async_ld_b128(lK0 + i * 512, gK + i * 512);
        async_ld_b128(lV0 + i * 512, gV + i * 512);
      }
      wait_async0();          // s_wait_asynccnt 0
      __syncthreads();
    }
    // ---- S = q kchunk^T  (fragments served by ds_load_b128 from LDS)
    v8f s0 = {}, s1 = {};
    {
      v16bf b00 = frag_contig((const bf16_t*)&Ks[ln * DH_ + half * 8]);
      v16bf b01 = frag_contig((const bf16_t*)&Ks[ln * DH_ + 32 + half * 8]);
      s0 = wmma_bf16(aq0, b00, s0);
      s0 = wmma_bf16(aq1, b01, s0);
      v16bf b10 = frag_contig((const bf16_t*)&Ks[(16 + ln) * DH_ + half * 8]);
      v16bf b11 = frag_contig((const bf16_t*)&Ks[(16 + ln) * DH_ + 32 + half * 8]);
      s1 = wmma_bf16(aq0, b10, s1);
      s1 = wmma_bf16(aq1, b11, s1);
    }
    // ---- online softmax update (row lives in 16 lanes of one half)
#pragma unroll
    for (int r = 0; r < 8; ++r) {
      float mx = fmaxf(s0[r], s1[r]);
#pragma unroll
      for (int off = 1; off < 16; off <<= 1) mx = fmaxf(mx, __shfl_xor(mx, off, 16));
      const float mnew = fmaxf(mrow[r], mx);
      const float p0 = __expf(s0[r] - mnew);
      const float p1 = __expf(s1[r] - mnew);
      float rs = p0 + p1;
#pragma unroll
      for (int off = 1; off < 16; off <<= 1) rs += __shfl_xor(rs, off, 16);
      const float corr = __expf(mrow[r] - mnew);
      lrow[r] = lrow[r] * corr + rs;
      mrow[r] = mnew;
#pragma unroll
      for (int t = 0; t < 4; ++t) o[t][r] *= corr;
      const int row = r + half * 8;
      Pl[row * 32 + ln]      = (bf16_t)p0;
      Pl[row * 32 + 16 + ln] = (bf16_t)p1;
    }
    __syncthreads();  // D-layout -> A-layout re-shape of P through LDS
    v16bf pa = frag_contig((const bf16_t*)&Pl[ln * 32 + half * 8]);
#pragma unroll
    for (int t = 0; t < 4; ++t) {
      v16bf bv = frag_b_gather((const bf16_t*)Vs, DH_, t * 16 + ln, half);
      o[t] = wmma_bf16(pa, bv, o[t]);
    }
    __syncthreads();
  }
#pragma unroll
  for (int r = 0; r < 8; ++r) {
    const float inv = 1.f / lrow[r];
    const int m = m0 + r + half * 8;
#pragma unroll
    for (int t = 0; t < 4; ++t) out[(long)m * DH_ + t * 16 + ln] = o[t][r] * inv;
  }
}

// ----------------------------------------------------- small helper kernels
__global__ void k_transpose_bf16(const float* __restrict__ W, bf16_t* __restrict__ WT,
                                 int R, int C) {
  long i = (long)blockIdx.x * blockDim.x + threadIdx.x;
  if (i >= (long)R * C) return;
  const int r = (int)(i / C), c = (int)(i % C);
  WT[(long)c * R + r] = (bf16_t)W[i];
}

__global__ void k_softmax256(float* __restrict__ X) {
  __shared__ float red[256];
  const long r = blockIdx.x;
  const int t = threadIdx.x;
  const float val = X[(r << 8) + t];
  red[t] = val; __syncthreads();
  for (int s = 128; s > 0; s >>= 1) { if (t < s) red[t] = fmaxf(red[t], red[t + s]); __syncthreads(); }
  const float mx = red[0]; __syncthreads();
  const float e = __expf(val - mx);
  red[t] = e; __syncthreads();
  for (int s = 128; s > 0; s >>= 1) { if (t < s) red[t] += red[t + s]; __syncthreads(); }
  X[(r << 8) + t] = e / red[0];
}

__global__ void k_scal_init(unsigned* s) { if (threadIdx.x < 2) s[threadIdx.x] = 0u; }

// global max of row-abs-sums and col-abs-sums (atomicMax on uint works: sums>=0)
__global__ void k_absmax(const float* __restrict__ X, unsigned* __restrict__ scal) {
  __shared__ float red[256];
  const int bh = blockIdx.x >> 8, i = blockIdx.x & 255, t = threadIdx.x;
  const float* base = X + ((long)bh << 16);
  const float rv = fabsf(base[(i << 8) + t]);
  const float cv = fabsf(base[(t << 8) + i]);
  red[t] = rv; __syncthreads();
  for (int s = 128; s > 0; s >>= 1) { if (t < s) red[t] += red[t + s]; __syncthreads(); }
  const float rs = red[0]; __syncthreads();
  red[t] = cv; __syncthreads();
  for (int s = 128; s > 0; s >>= 1) { if (t < s) red[t] += red[t + s]; __syncthreads(); }
  if (t == 0) {
    atomicMax(&scal[0], __float_as_uint(rs));
    atomicMax(&scal[1], __float_as_uint(red[0]));
  }
}

__global__ void k_zinit(const float* __restrict__ A2, float* __restrict__ Z,
                        const unsigned* __restrict__ scal) {
  long i = (long)blockIdx.x * blockDim.x + threadIdx.x;
  if (i >= (long)BH_ * M_ * M_) return;
  const int c = (int)(i & 255), r = (int)((i >> 8) & 255);
  const long bh = i >> 16;
  const float inv = 1.f / (__uint_as_float(scal[0]) * __uint_as_float(scal[1]));
  Z[i] = A2[(bh << 16) + ((long)c << 8) + r] * inv;  // transpose
}

__global__ void k_diag_minus(float* __restrict__ D, const float* __restrict__ S, float dval) {
  long i = (long)blockIdx.x * blockDim.x + threadIdx.x;
  if (i >= (long)BH_ * M_ * M_) return;
  const int c = (int)(i & 255), r = (int)((i >> 8) & 255);
  D[i] = ((r == c) ? dval : 0.f) - S[i];
}

__global__ void k_f32_to_bf16(const float* __restrict__ S, bf16_t* __restrict__ D, long total) {
  long i = (long)blockIdx.x * blockDim.x + threadIdx.x;
  if (i < total) D[i] = (bf16_t)S[i];
}

// depthwise 33-tap conv over sequence (v) + attention output -> y bf16 [b,n,h*dh]
__global__ void k_conv_combine(const float* __restrict__ OA, const bf16_t* __restrict__ Vb,
                               const float* __restrict__ rk, bf16_t* __restrict__ Y) {
  long i = (long)blockIdx.x * blockDim.x + threadIdx.x;
  if (i >= (long)BH_ * N_ * DH_) return;
  const int d = (int)(i & 63);
  const int nidx = (int)((i >> 6) & (N_ - 1));
  const long bh = i >> 19;
  const int h = (int)(bh & 7);
  const long bb = bh >> 3;
  float acc = OA[i];
  const bf16_t* vv = Vb + (bh * N_) * DH_ + d;
#pragma unroll
  for (int t = 0; t < 33; ++t) {
    const int s = nidx + t - 16;
    if (s >= 0 && s < N_) acc += rk[h * 33 + t] * (float)vv[(long)s * DH_];
  }
  Y[(bb * N_ + nidx) * INNER_ + (long)h * DH_ + d] = (bf16_t)acc;
}

// =========================================================== orchestration
extern "C" void kernel_launch(void* const* d_in, const int* in_sizes, int n_in,
                              void* d_out, int out_size, void* d_ws, size_t ws_size,
                              hipStream_t stream) {
  const float* x    = (const float*)d_in[0];  // [4,8192,512]
  const float* wqkv = (const float*)d_in[1];  // [512,1536]
  const float* wout = (const float*)d_in[2];  // [512,512]
  const float* bout = (const float*)d_in[3];  // [512]
  const float* rk   = (const float*)d_in[4];  // [8,1,33,1]
  (void)in_sizes; (void)n_in; (void)out_size; (void)ws_size;

  char* base = (char*)d_ws;
  size_t off = 0;
  auto alloc = [&](size_t bytes) -> void* {
    off = (off + 255) & ~(size_t)255;
    void* p = base + off;
    off += bytes;
    return p;
  };
  unsigned* scal  = (unsigned*)alloc(8);
  bf16_t* wqkvT = (bf16_t*)alloc((size_t)KQKV_ * DIM_ * 2);
  bf16_t* woutT = (bf16_t*)alloc((size_t)DIM_ * INNER_ * 2);
  bf16_t* Qb    = (bf16_t*)alloc((size_t)BH_ * N_ * DH_ * 2);
  bf16_t* Kb    = (bf16_t*)alloc((size_t)BH_ * N_ * DH_ * 2);
  bf16_t* Vb    = (bf16_t*)alloc((size_t)BH_ * N_ * DH_ * 2);
  bf16_t* Ql    = (bf16_t*)alloc((size_t)BH_ * M_ * DH_ * 2);
  bf16_t* Kl    = (bf16_t*)alloc((size_t)BH_ * M_ * DH_ * 2);
  float*  A2f   = (float*)alloc((size_t)BH_ * M_ * M_ * 4);
  float*  ZA    = (float*)alloc((size_t)BH_ * M_ * M_ * 4);
  float*  ZB    = (float*)alloc((size_t)BH_ * M_ * M_ * 4);
  float*  XZ    = (float*)alloc((size_t)BH_ * M_ * M_ * 4);
  float*  T1    = (float*)alloc((size_t)BH_ * M_ * M_ * 4);
  float*  T2    = (float*)alloc((size_t)BH_ * M_ * M_ * 4);
  float*  A3V   = (float*)alloc((size_t)BH_ * M_ * DH_ * 4);
  float*  Wf    = (float*)alloc((size_t)BH_ * M_ * DH_ * 4);
  bf16_t* Wb    = (bf16_t*)alloc((size_t)BH_ * M_ * DH_ * 2);
  float*  OA    = (float*)alloc((size_t)BH_ * N_ * DH_ * 4);
  bf16_t* Y     = (bf16_t*)alloc((size_t)ROWS_ * INNER_ * 2);

  const long sLand = (long)M_ * DH_;    // 16384
  const long sMM   = (long)M_ * M_;     // 65536

  // 1) weights -> transposed bf16 (contiguous-K fast path for all big GEMMs)
  k_transpose_bf16<<<(DIM_ * KQKV_ + 255) / 256, 256, 0, stream>>>(wqkv, wqkvT, DIM_, KQKV_);
  k_transpose_bf16<<<(DIM_ * INNER_ + 255) / 256, 256, 0, stream>>>(wout, woutT, DIM_, INNER_);

  // 2) qkv projection + head split/scale (WMMA)
  k_qkv_gemm<<<dim3(KQKV_ / 64, ROWS_ / 16), 32, 0, stream>>>(x, wqkvT, Qb, Kb, Vb);

  // 3) landmark means
  k_landmarks<<<(BH_ * M_ * DH_ + 255) / 256, 256, 0, stream>>>(Qb, Kb, Ql, Kl);

  // 4) attn2 = softmax(q_land k_land^T)   [bh,256,256]
  gemm_wmma<bf16_t, bf16_t, true, false><<<dim3(M_ / 64, M_ / 16, BH_), 32, 0, stream>>>(
      Ql, Kl, A2f, M_, M_, DH_, DH_, DH_, M_, sLand, sLand, sMM, 1.f, 0.f, nullptr);
  k_softmax256<<<BH_ * M_, 256, 0, stream>>>(A2f);

  // 5) attn3 @ v  (fused flash over n=8192)   [bh,256,64]
  k_flash16<<<dim3(M_ / 16, BH_), 32, 0, stream>>>(Ql, Kb, Vb, A3V, M_, N_);

  // 6) Moore-Penrose pinv of attn2 (6 Newton-Schulz iterations)
  k_scal_init<<<1, 32, 0, stream>>>(scal);
  k_absmax<<<BH_ * M_, 256, 0, stream>>>(A2f, scal);
  k_zinit<<<(BH_ * M_ * M_ + 255) / 256, 256, 0, stream>>>(A2f, ZA, scal);
  float* zc = ZA;
  float* zn = ZB;
  const dim3 gMM(M_ / 64, M_ / 16, BH_);
  for (int it = 0; it < 6; ++it) {
    // XZ = attn2 @ z
    gemm_wmma<float, float, false, false><<<gMM, 32, 0, stream>>>(
        A2f, zc, XZ, M_, M_, M_, M_, M_, M_, sMM, sMM, sMM, 1.f, 0.f, nullptr);
    // T1 = 7I - XZ
    k_diag_minus<<<(BH_ * M_ * M_ + 255) / 256, 256, 0, stream>>>(T1, XZ, 7.f);
    // T2 = 15I - XZ @ T1
    gemm_wmma<float, float, false, false><<<gMM, 32, 0, stream>>>(
        XZ, T1, T2, M_, M_, M_, M_, M_, M_, sMM, sMM, sMM, -1.f, 15.f, nullptr);
    // T1 = 13I - XZ @ T2
    gemm_wmma<float, float, false, false><<<gMM, 32, 0, stream>>>(
        XZ, T2, T1, M_, M_, M_, M_, M_, M_, sMM, sMM, sMM, -1.f, 13.f, nullptr);
    // z' = 0.25 * z @ T1
    gemm_wmma<float, float, false, false><<<gMM, 32, 0, stream>>>(
        zc, T1, zn, M_, M_, M_, M_, M_, M_, sMM, sMM, sMM, 0.25f, 0.f, nullptr);
    float* tmp = zc; zc = zn; zn = tmp;
  }

  // 7) W = attn2_inv @ (attn3 @ v)   [bh,256,64]
  gemm_wmma<float, float, false, false><<<dim3(1, M_ / 16, BH_), 32, 0, stream>>>(
      zc, A3V, Wf, M_, DH_, M_, M_, DH_, DH_, sMM, sLand, sLand, 1.f, 0.f, nullptr);
  k_f32_to_bf16<<<(BH_ * M_ * DH_ + 255) / 256, 256, 0, stream>>>(Wf, Wb, (long)BH_ * M_ * DH_);

  // 8) out = softmax(q k_land^T) @ W  (fused flash over m=256)  [bh,8192,64]
  k_flash16<<<dim3(N_ / 16, BH_), 32, 0, stream>>>(Qb, Kl, Wb, OA, N_, M_);

  // 9) + depthwise residual conv on v, transpose to [b,n,h*dh] bf16
  k_conv_combine<<<(BH_ * N_ * DH_ + 255) / 256, 256, 0, stream>>>(OA, Vb, rk, Y);

  // 10) final projection + bias -> d_out f32 [b,n,512]
  gemm_wmma<bf16_t, bf16_t, true, true><<<dim3(INNER_ / 64, ROWS_ / 16, 1), 32, 0, stream>>>(
      Y, woutT, (float*)d_out, ROWS_, DIM_, INNER_, INNER_, INNER_, DIM_,
      0L, 0L, 0L, 1.f, 0.f, bout);
}
